// MixtureOfExperts_27814208208932
// MI455X (gfx1250) — matrix operand order repaired and checked by reading, not compiled, not run on hardware
//
#include <hip/hip_runtime.h>
#include <hip/hip_bf16.h>
#include <stdint.h>

// ---------------------------------------------------------------------------
// MoE on gfx1250: bf16 WMMA (v_wmma_f32_16x16x32_bf16) GEMM core with
// double-buffered LDS filled by global_load_async_to_lds_b128 (ASYNCcnt).
// K-loop unrolled over both LDS stages; fragments batched before WMMA issue.
// B=4096, IN=1024, HID=4096, OUT=1024, E=8  (all fp32 in/out)
// ---------------------------------------------------------------------------

typedef __attribute__((ext_vector_type(16))) __bf16 v16bf;
typedef __attribute__((ext_vector_type(8)))  __bf16 v8bf;
typedef __attribute__((ext_vector_type(8)))  float  v8f;

#define MOE_B    4096
#define MOE_IN   1024
#define MOE_HID  4096
#define MOE_OUT  1024
#define MOE_E    8

// ---------------------------------------------------------------------------
// CDNA5 async copy: 16B global -> LDS per lane, tracked by ASYNCcnt.
// ---------------------------------------------------------------------------
__device__ inline void async_ld16(uint32_t lds_byte_off, const void* gaddr) {
    asm volatile("global_load_async_to_lds_b128 %0, %1, off"
                 :
                 : "v"(lds_byte_off), "v"(gaddr)
                 : "memory");
}
__device__ inline void wait_async0() {
    asm volatile("s_wait_asynccnt 0x0" ::: "memory");
}

// ---------------------------------------------------------------------------
// f32 -> bf16 elementwise cast (x activations; layout preserved, M x K)
// ---------------------------------------------------------------------------
__global__ __launch_bounds__(256) void cast_f32_bf16(const float* __restrict__ in,
                                                     __bf16* __restrict__ out, int n) {
    int i = (blockIdx.x * 256 + threadIdx.x) * 4;
    if (i + 3 < n) {
        float4 v = *(const float4*)(in + i);
        out[i + 0] = (__bf16)v.x;
        out[i + 1] = (__bf16)v.y;
        out[i + 2] = (__bf16)v.z;
        out[i + 3] = (__bf16)v.w;
    }
}

// ---------------------------------------------------------------------------
// Transpose-convert: in f32 [R][C]  ->  out bf16 [C][R]   (weights -> Bt[N][K])
// R, C multiples of 32.
// ---------------------------------------------------------------------------
__global__ __launch_bounds__(256) void transpose_f32_bf16(const float* __restrict__ in,
                                                          __bf16* __restrict__ out,
                                                          int R, int C) {
    __shared__ float tile[32][33];
    int tx = threadIdx.x;          // 0..31
    int ty = threadIdx.y;          // 0..7
    int c0 = blockIdx.x * 32;
    int r0 = blockIdx.y * 32;
#pragma unroll
    for (int j = 0; j < 32; j += 8)
        tile[ty + j][tx] = in[(size_t)(r0 + ty + j) * C + (c0 + tx)];
    __syncthreads();
#pragma unroll
    for (int j = 0; j < 32; j += 8)
        out[(size_t)(c0 + ty + j) * R + (r0 + tx)] = (__bf16)tile[tx][ty + j];
}

// ---------------------------------------------------------------------------
// WMMA fragment builders from LDS (16-bit operand layouts, CDNA5 ISA 7.12.2)
// A (16x32): lane half selects K-chunks {khalf*8..+7} and {16+khalf*8..+7}
// B (32x16): lane half selects K-chunks {khalf*8..+7} and {16+khalf*8..+7}
// ---------------------------------------------------------------------------
__device__ inline v16bf make_afrag(const __bf16* rowbase, int khalf) {
    v8bf lo = *(const v8bf*)(rowbase + khalf * 8);
    v8bf hi = *(const v8bf*)(rowbase + 16 + khalf * 8);
    return __builtin_shufflevector(lo, hi, 0, 1, 2, 3, 4, 5, 6, 7,
                                           8, 9, 10, 11, 12, 13, 14, 15);
}
__device__ inline v16bf make_bfrag(const __bf16* rowbase, int khalf) {
    v8bf lo = *(const v8bf*)(rowbase + khalf * 8);
    v8bf hi = *(const v8bf*)(rowbase + 16 + khalf * 8);
    return __builtin_shufflevector(lo, hi, 0, 1, 2, 3, 4, 5, 6, 7,
                                           8, 9, 10, 11, 12, 13, 14, 15);
}

// ---------------------------------------------------------------------------
// Tiled bf16 WMMA GEMM:  C[M][N] = A[M][K] * Bt[N][K]^T  (+bias epilogues)
// Block = 256 thr (8 waves, 2x4), tile 128x128, K-step 32, 2 LDS stages,
// K-loop unrolled over both stages (K must be a multiple of 64).
// MODE 0: H[row][col] = bf16( relu(acc + bias[col]) )
// MODE 1: Fout[row][col]  = gate[row*8+e] * (acc + bias[col])      (init)
// MODE 2: Fout[row][col] += gate[row*8+e] * (acc + bias[col])      (accum)
// ---------------------------------------------------------------------------
template <int MODE>
__global__ __launch_bounds__(256) void gemm_bf16_wmma(
    const __bf16* __restrict__ A,     // [M][K]
    const __bf16* __restrict__ Bt,    // [N][K]
    const float*  __restrict__ bias,  // [N]
    __bf16*       __restrict__ Hout,  // MODE 0
    float*        __restrict__ Fout,  // MODE 1/2
    const float*  __restrict__ gate,  // [M][8], MODE 1/2
    int expert, int M, int N, int K) {

    __shared__ __align__(16) __bf16 lA[2][128 * 32];
    __shared__ __align__(16) __bf16 lB[2][128 * 32];

    const int tid  = threadIdx.x;
    const int lane = tid & 31;
    const int wid  = tid >> 5;
    const int wm   = wid >> 2;   // 0..1  (64 rows each)
    const int wn   = wid & 3;    // 0..3  (32 cols each)
    const int rowBase = blockIdx.y * 128;
    const int colBase = blockIdx.x * 128;

    const int khalf = lane >> 4;
    const int l16   = lane & 15;

    v8f acc[4][2];
#pragma unroll
    for (int mt = 0; mt < 4; mt++)
#pragma unroll
        for (int nt = 0; nt < 2; nt++)
#pragma unroll
            for (int r = 0; r < 8; r++) acc[mt][nt][r] = 0.0f;

    // global->LDS copy mapping: 512 chunks of 8 bf16 (16B); 2 chunks per thread
    const int cA0 = tid;
    const int cA1 = tid + 256;
    const int rA0 = cA0 >> 2, oA0 = (cA0 & 3) * 8;
    const int rA1 = cA1 >> 2, oA1 = (cA1 & 3) * 8;

    const uint32_t ldsA0[2] = {(uint32_t)(uintptr_t)&lA[0][rA0 * 32 + oA0],
                               (uint32_t)(uintptr_t)&lA[1][rA0 * 32 + oA0]};
    const uint32_t ldsA1[2] = {(uint32_t)(uintptr_t)&lA[0][rA1 * 32 + oA1],
                               (uint32_t)(uintptr_t)&lA[1][rA1 * 32 + oA1]};
    const uint32_t ldsB0[2] = {(uint32_t)(uintptr_t)&lB[0][rA0 * 32 + oA0],
                               (uint32_t)(uintptr_t)&lB[1][rA0 * 32 + oA0]};
    const uint32_t ldsB1[2] = {(uint32_t)(uintptr_t)&lB[0][rA1 * 32 + oA1],
                               (uint32_t)(uintptr_t)&lB[1][rA1 * 32 + oA1]};

    const __bf16* gA0 = &A [(size_t)(rowBase + rA0) * K + oA0];
    const __bf16* gA1 = &A [(size_t)(rowBase + rA1) * K + oA1];
    const __bf16* gB0 = &Bt[(size_t)(colBase + rA0) * K + oA0];
    const __bf16* gB1 = &Bt[(size_t)(colBase + rA1) * K + oA1];

    auto issue = [&](int stg, int kk) {
        async_ld16(ldsA0[stg], gA0 + kk);
        async_ld16(ldsA1[stg], gA1 + kk);
        async_ld16(ldsB0[stg], gB0 + kk);
        async_ld16(ldsB1[stg], gB1 + kk);
    };

    auto compute = [&](const __bf16* baseA, const __bf16* baseB) {
        v16bf bfrag[2], afrag[4];
#pragma unroll
        for (int nt = 0; nt < 2; nt++)
            bfrag[nt] = make_bfrag(baseB + (wn * 32 + nt * 16 + l16) * 32, khalf);
#pragma unroll
        for (int mt = 0; mt < 4; mt++)
            afrag[mt] = make_afrag(baseA + (wm * 64 + mt * 16 + l16) * 32, khalf);
#pragma unroll
        for (int mt = 0; mt < 4; mt++)
#pragma unroll
            for (int nt = 0; nt < 2; nt++)
                acc[mt][nt] = __builtin_amdgcn_wmma_f32_16x16x32_bf16(
                    false, afrag[mt], false, bfrag[nt], (short)0, acc[mt][nt],
                    false, false);
    };

    // prologue: stage 0
    issue(0, 0);

    for (int k0 = 0; k0 < K; k0 += 64) {
        // ---- stage 0: K chunk [k0, k0+32) ----
        wait_async0();        // stage 0 resident in LDS
        __syncthreads();      // visible to all; stage-1 reads drained
        issue(1, k0 + 32);    // K multiple of 64 -> always valid
        compute(lA[0], lB[0]);
        __syncthreads();      // stage-0 reads drained before refill

        // ---- stage 1: K chunk [k0+32, k0+64) ----
        wait_async0();
        __syncthreads();
        if (k0 + 64 < K) issue(0, k0 + 64);
        compute(lA[1], lB[1]);
        __syncthreads();
    }

    // Epilogue.  C/D layout: lanes 0-15 -> M=r, lanes 16-31 -> M=8+r; N=lane&15
    const int mh = (lane >> 4) * 8;
#pragma unroll
    for (int mt = 0; mt < 4; mt++) {
#pragma unroll
        for (int nt = 0; nt < 2; nt++) {
            const int col = colBase + wn * 32 + nt * 16 + l16;
            const float bv = bias[col];
#pragma unroll
            for (int r = 0; r < 8; r++) {
                const int row = rowBase + wm * 64 + mt * 16 + mh + r;
                float v = acc[mt][nt][r] + bv;
                if (MODE == 0) {
                    v = v > 0.0f ? v : 0.0f;
                    Hout[(size_t)row * N + col] = (__bf16)v;
                } else {
                    const float g = gate[row * 8 + expert];
                    v *= g;
                    if (MODE == 1)
                        Fout[(size_t)row * N + col] = v;
                    else
                        Fout[(size_t)row * N + col] += v;
                }
            }
        }
    }
}

// ---------------------------------------------------------------------------
// Gating logits + softmax: probs[row][e] = softmax_e( H[row]·gW2[:,e] + gb2 )
// One block per row.
// ---------------------------------------------------------------------------
__global__ __launch_bounds__(256) void gating_logits_softmax(
    const __bf16* __restrict__ H, const float* __restrict__ gW2,
    const float* __restrict__ gb2, float* __restrict__ probs, int HID) {
    __shared__ float red[256 * 8];
    const int row = blockIdx.x;
    const int tid = threadIdx.x;
    float p[8];
#pragma unroll
    for (int e = 0; e < 8; e++) p[e] = 0.0f;
    for (int k = tid; k < HID; k += 256) {
        const float hv = (float)H[(size_t)row * HID + k];
#pragma unroll
        for (int e = 0; e < 8; e++) p[e] += hv * gW2[k * 8 + e];
    }
#pragma unroll
    for (int e = 0; e < 8; e++) red[tid * 8 + e] = p[e];
    for (int s = 128; s > 0; s >>= 1) {
        __syncthreads();
        if (tid < s)
#pragma unroll
            for (int e = 0; e < 8; e++) red[tid * 8 + e] += red[(tid + s) * 8 + e];
    }
    __syncthreads();
    if (tid == 0) {
        float l[8], mx = -3.4e38f;
#pragma unroll
        for (int e = 0; e < 8; e++) { l[e] = red[e] + gb2[e]; mx = l[e] > mx ? l[e] : mx; }
        float sum = 0.0f;
#pragma unroll
        for (int e = 0; e < 8; e++) { l[e] = __expf(l[e] - mx); sum += l[e]; }
        const float inv = 1.0f / sum;
#pragma unroll
        for (int e = 0; e < 8; e++) probs[row * 8 + e] = l[e] * inv;
    }
}

// ---------------------------------------------------------------------------
// Sequential gating scan (reference's lax.scan over rows). Single wave32;
// lanes 0..7 carry the 8-expert state, width-8 shuffle reductions.
// Writes gate_ws [B][8] and the `g` tail of d_out.
// ---------------------------------------------------------------------------
__global__ void gating_scan(const float* __restrict__ probs,
                            float* __restrict__ gate_ws,
                            float* __restrict__ g_out, int B) {
    const int lane = threadIdx.x;   // 32 threads
    float rta = 0.0f;
    int any_nonzero = 0;
    for (int i = 0; i < B; i++) {
        const float s = (lane < 8) ? probs[i * 8 + lane] : 0.0f;
        const float rta1 = rta + s;
        float sum = rta1;
        sum += __shfl_xor(sum, 1, 8);
        sum += __shfl_xor(sum, 2, 8);
        sum += __shfl_xor(sum, 4, 8);
        const float mean = sum * 0.125f;
        int mask = (rta1 > mean + 0.1f) ? 1 : 0;
        int anym = mask;
        anym |= __shfl_xor(anym, 1, 8);
        anym |= __shfl_xor(anym, 2, 8);
        anym |= __shfl_xor(anym, 4, 8);
        const float rta2 = anym ? (rta1 - s) : rta1;
        float mod = mask ? 0.0f : s;
        float norm = mod;
        norm += __shfl_xor(norm, 1, 8);
        norm += __shfl_xor(norm, 2, 8);
        norm += __shfl_xor(norm, 4, 8);
        mod = mod / ((norm == 0.0f) ? 1.0f : norm);
        if (lane < 8) {
            gate_ws[i * 8 + lane] = mod;
            g_out[i * 8 + lane]   = mod;
            if (mod != 0.0f) any_nonzero = 1;
        }
        rta = rta2 + mod;
    }
    any_nonzero |= __shfl_xor(any_nonzero, 1, 8);
    any_nonzero |= __shfl_xor(any_nonzero, 2, 8);
    any_nonzero |= __shfl_xor(any_nonzero, 4, 8);
    if (!any_nonzero && lane < 8) {
        for (int i = 0; i < B; i++) {
            gate_ws[i * 8 + lane] = 0.125f;
            g_out[i * 8 + lane]   = 0.125f;
        }
    }
}

// ---------------------------------------------------------------------------
// Host-side orchestration
// ---------------------------------------------------------------------------
extern "C" void kernel_launch(void* const* d_in, const int* in_sizes, int n_in,
                              void* d_out, int out_size, void* d_ws, size_t ws_size,
                              hipStream_t stream) {
    (void)in_sizes; (void)n_in; (void)out_size; (void)ws_size;

    const float* x    = (const float*)d_in[0];  // [B][IN]
    const float* gW1  = (const float*)d_in[1];  // [IN][HID]
    const float* gb1  = (const float*)d_in[2];  // [HID]
    const float* gW2  = (const float*)d_in[3];  // [HID][E]
    const float* gb2  = (const float*)d_in[4];  // [E]
    const float* eW1  = (const float*)d_in[5];  // [E][IN][HID]
    const float* eb1  = (const float*)d_in[6];  // [E][HID]
    const float* eW2  = (const float*)d_in[7];  // [E][HID][OUT]
    const float* eb2  = (const float*)d_in[8];  // [E][OUT]

    float* out   = (float*)d_out;                              // [B][OUT]
    float* g_out = (float*)d_out + (size_t)MOE_B * MOE_OUT;    // [B][E]

    // workspace layout (bytes)
    char* ws = (char*)d_ws;
    __bf16* xb    = (__bf16*)(ws);                               // 8 MB  [B][IN]
    __bf16* Wt    = (__bf16*)(ws + (8u << 20));                  // 8 MB  [N][K] (reused)
    __bf16* H     = (__bf16*)(ws + (16u << 20));                 // 32 MB [B][HID]
    float*  probs = (float*)(ws + (48u << 20));                  // 128 KB
    float*  gate  = (float*)(ws + (48u << 20) + (128u << 10));   // 128 KB

    const dim3 blk256(256);
    const dim3 blkT(32, 8);

    // 1) x -> bf16
    cast_f32_bf16<<<dim3((MOE_B * MOE_IN) / (256 * 4)), blk256, 0, stream>>>(
        x, xb, MOE_B * MOE_IN);

    // 2) gating GEMM1: H = relu(x @ gW1 + gb1)
    transpose_f32_bf16<<<dim3(MOE_HID / 32, MOE_IN / 32), blkT, 0, stream>>>(
        gW1, Wt, MOE_IN, MOE_HID);
    gemm_bf16_wmma<0><<<dim3(MOE_HID / 128, MOE_B / 128), blk256, 0, stream>>>(
        xb, Wt, gb1, H, nullptr, nullptr, 0, MOE_B, MOE_HID, MOE_IN);

    // 3) logits + softmax -> probs
    gating_logits_softmax<<<dim3(MOE_B), blk256, 0, stream>>>(H, gW2, gb2, probs,
                                                              MOE_HID);

    // 4) sequential scan -> gate (and g tail of d_out)
    gating_scan<<<dim3(1), dim3(32), 0, stream>>>(probs, gate, g_out, MOE_B);

    // 5) experts
    for (int e = 0; e < MOE_E; e++) {
        const float* W1e = eW1 + (size_t)e * MOE_IN * MOE_HID;
        const float* b1e = eb1 + (size_t)e * MOE_HID;
        const float* W2e = eW2 + (size_t)e * MOE_HID * MOE_OUT;
        const float* b2e = eb2 + (size_t)e * MOE_OUT;

        transpose_f32_bf16<<<dim3(MOE_HID / 32, MOE_IN / 32), blkT, 0, stream>>>(
            W1e, Wt, MOE_IN, MOE_HID);
        gemm_bf16_wmma<0><<<dim3(MOE_HID / 128, MOE_B / 128), blk256, 0, stream>>>(
            xb, Wt, b1e, H, nullptr, nullptr, 0, MOE_B, MOE_HID, MOE_IN);

        transpose_f32_bf16<<<dim3(MOE_OUT / 32, MOE_HID / 32), blkT, 0, stream>>>(
            W2e, Wt, MOE_HID, MOE_OUT);
        if (e == 0) {
            gemm_bf16_wmma<1><<<dim3(MOE_OUT / 128, MOE_B / 128), blk256, 0, stream>>>(
                H, Wt, b2e, nullptr, out, gate, e, MOE_B, MOE_OUT, MOE_HID);
        } else {
            gemm_bf16_wmma<2><<<dim3(MOE_OUT / 128, MOE_B / 128), blk256, 0, stream>>>(
                H, Wt, b2e, nullptr, out, gate, e, MOE_B, MOE_OUT, MOE_HID);
        }
    }
}
